// GuitarToPianoResynthModel_77446850282099
// MI455X (gfx1250) — matrix operand order repaired and checked by reading, not compiled, not run on hardware
//
#include <hip/hip_runtime.h>

// ---------------------------------------------------------------------------
// GuitarToPianoResynthModel — CDNA5 (gfx1250) implementation.
//  * convs + GRU input projections: implicit GEMM on v_wmma_f32_16x16x32_bf16,
//    64x128 macro tile, bf16 weights pre-converted once per launch
//  * A-tile staging via async global->LDS (ASYNCcnt) when available
//  * GroupNorm stats fused into the conv epilogue (deterministic tree reduce)
//  * GRU recurrence: one persistent 24-wave workgroup, WMMA per time step
// ---------------------------------------------------------------------------

typedef __attribute__((ext_vector_type(16))) __bf16 v16bf;
typedef __attribute__((ext_vector_type(8)))  __bf16 bf16x8;
typedef __attribute__((ext_vector_type(8)))  unsigned short u16x8;
typedef __attribute__((ext_vector_type(8)))  float v8f;
typedef __attribute__((ext_vector_type(4)))  int v4i;

#if __has_builtin(__builtin_amdgcn_global_load_async_to_lds_b128)
#define HAVE_ASYNC_LDS 1
#else
#define HAVE_ASYNC_LDS 0
#endif

#define DEV __device__ __forceinline__

DEV unsigned short f2bf(float f) {          // f32 -> bf16, round-nearest-even
  unsigned int u = __float_as_uint(f);
  u += 0x7fffu + ((u >> 16) & 1u);
  return (unsigned short)(u >> 16);
}
DEV float geluf(float x) { return 0.5f * x * (1.0f + erff(x * 0.70710678118654752f)); }
DEV float sigmf(float x) { return 1.0f / (1.0f + expf(-x)); }
DEV bf16x8 ld8(const unsigned short* p) { return __builtin_bit_cast(bf16x8, *(const u16x8*)p); }
DEV v16bf cat16(bf16x8 lo, bf16x8 hi) {
  return __builtin_shufflevector(lo, hi, 0,1,2,3,4,5,6,7,8,9,10,11,12,13,14,15);
}
DEV void wait_async_lds() {
#if HAVE_ASYNC_LDS
#if __has_builtin(__builtin_amdgcn_s_wait_asynccnt)
  __builtin_amdgcn_s_wait_asynccnt(0);
#else
  asm volatile("s_wait_asynccnt 0x0" ::: "memory");
#endif
#endif
}

// ------------------------- concat (x_ctx, onset_ctx) -----------------------
__global__ void concat2_kernel(const float* __restrict__ a, const float* __restrict__ b,
                               float* __restrict__ y, int L) {
  int i = blockIdx.x * blockDim.x + threadIdx.x;   // over 16*L
  int bb = i / L, l = i - bb * L;
  y[((size_t)bb * 2 + 0) * L + l] = a[i];
  y[((size_t)bb * 2 + 1) * L + l] = b[i];
}

// ------------------------- f32 -> bf16 convert -----------------------------
__global__ void cvt_bf16_kernel(const float* __restrict__ x, unsigned short* __restrict__ y, int n) {
  int i = blockIdx.x * blockDim.x + threadIdx.x;
  if (i < n) y[i] = f2bf(x[i]);
}

// ---------------------------------------------------------------------------
// Causal dilated conv as implicit GEMM (bf16 weights, f32 accumulate).
//   Y[b,co,l] = bias[co] + sum_{ci,t} W[co,ci,t] * X[b,ci, l*stride-(k-1-t)*dil]
// M=Cout, N=l (per batch), K=Cin*k.  Macro tile 64x128, BK=32.
// 8 waves: wm=wid&3 (m-tile), wn=wid>>2 (64-col half); 4 WMMAs/wave/chunk.
// act: 0=none, 1=exact GELU.  part!=null: emit per-(b,group,lblk) sum/sumsq.
// ---------------------------------------------------------------------------
#define BK_PAD 40   // halfs per LDS row (80B, 16B-aligned)

__global__ __launch_bounds__(256) void conv_wmma_kernel(
    const float* __restrict__ X, const unsigned short* __restrict__ Wbf,
    const float* __restrict__ bias, float* __restrict__ Y, float* part,
    int Cin, int Cout, int Lout, int Lin, int ksize, int stride, int dil,
    int act, int Cg) {
  __shared__ __align__(16) unsigned short As[64 * BK_PAD];
  __shared__ __align__(16) unsigned short Bs[128 * BK_PAD];
  __shared__ float wv_s[8], wv_q[8];

  const int tid  = threadIdx.x;
  const int lane = tid & 31;
  const int wid  = tid >> 5;
  const int wm   = wid & 3;          // m-tile 0..3
  const int wn   = wid >> 2;         // n-half 0..1 (64 cols each)
  const int b    = blockIdx.z;
  const int l0   = blockIdx.x * 128;
  const int m0   = blockIdx.y * 64;
  const int K    = Cin * ksize;
  const int nchunk = (K + 31) >> 5;

  const int half8 = (lane & 16) ? 8 : 0;   // A-frag K sub-offset
  const int klo16 = (lane & 16);           // added then subtracted in B addressing; B-frag uses half8
  const int mrowA = wm * 16 + (lane & 15);

  // staging assignments
  const int arow = tid >> 2;               // A row 0..63
  const int akoct = (tid & 3) * 8;         // A K octet
  const int brow = tid >> 1;               // B row 0..127
  const int bk16 = (tid & 1) * 16;         // B K 16-run
  const unsigned short* Wrow = Wbf + (size_t)(m0 + arow) * K;
  const int lstg = l0 + brow;
  const bool avec = ((K & 7) == 0);        // vector/async path legal (alignment)

  v8f acc[4]; acc[0] = {}; acc[1] = {}; acc[2] = {}; acc[3] = {};

  for (int ch = 0; ch < nchunk; ++ch) {
    const int k0 = ch * 32;
    // ---- stage A (bf16 weights, bulk copy; async when available) ----
    unsigned short* adst = &As[arow * BK_PAD + akoct];
    if (avec && (k0 + akoct + 8 <= K)) {
#if HAVE_ASYNC_LDS
      __builtin_amdgcn_global_load_async_to_lds_b128(
          (v4i*)(Wrow + k0 + akoct), (v4i*)adst, 0, 0);
#else
      *(u16x8*)adst = *(const u16x8*)(Wrow + k0 + akoct);
#endif
    } else {
#pragma unroll
      for (int i = 0; i < 8; ++i) {
        int gk = k0 + akoct + i;
        adst[i] = (gk < K) ? Wrow[gk] : (unsigned short)0;
      }
    }
    // ---- stage B (im2col gather, f32 -> bf16) ----
#pragma unroll
    for (int i = 0; i < 16; ++i) {
      int gk = k0 + bk16 + i;
      float v = 0.0f;
      if (gk < K) {
        int ci = gk / ksize;
        int t  = gk - ci * ksize;
        int li = lstg * stride - (ksize - 1 - t) * dil;
        if (li >= 0 && li < Lin) v = X[((size_t)b * Cin + ci) * Lin + li];
      }
      Bs[brow * BK_PAD + bk16 + i] = f2bf(v);
    }
    wait_async_lds();
    __syncthreads();
    // ---- fragments + WMMA ----
    const unsigned short* pa = As + mrowA * BK_PAD;
    v16bf afrag = cat16(ld8(pa + half8), ld8(pa + 16 + half8));
#pragma unroll
    for (int sub = 0; sub < 4; ++sub) {
      const unsigned short* pb = Bs + (wn * 64 + sub * 16 + (lane & 15)) * BK_PAD + klo16;
      v16bf bfrag = cat16(ld8(pb - klo16 + half8), ld8(pb - klo16 + 16 + half8));
      acc[sub] = __builtin_amdgcn_wmma_f32_16x16x32_bf16(false, afrag, false, bfrag,
                                                         (short)0, acc[sub], false, false);
    }
    __syncthreads();
  }
  // ---- epilogue: D layout m = r + 8*(lane>=16), n = lane&15 ----
  const int mhi = (lane & 16) ? 8 : 0;
  const int nl  = lane & 15;
  float ssum = 0.f, qsum = 0.f;
#pragma unroll
  for (int r = 0; r < 8; ++r) {
    int co = m0 + wm * 16 + mhi + r;
    float bv = bias[co];
    float* yrow = Y + ((size_t)b * Cout + co) * Lout;
#pragma unroll
    for (int sub = 0; sub < 4; ++sub) {
      float v = acc[sub][r] + bv;
      if (act == 1) v = geluf(v);
      yrow[l0 + wn * 64 + sub * 16 + nl] = v;
      ssum += v; qsum += v * v;
    }
  }
  // ---- fused GroupNorm partial stats (deterministic tree reduce) ----
  if (part != nullptr) {
    // all of a wave's outputs sit in one group (tiles are 16-aligned, Cg>=16)
#pragma unroll
    for (int o = 16; o > 0; o >>= 1) {
      ssum += __shfl_down(ssum, o);
      qsum += __shfl_down(qsum, o);
    }
    if (lane == 0) { wv_s[wid] = ssum; wv_q[wid] = qsum; }
    __syncthreads();
    int ngrp = 64 / Cg;                    // groups covered by this block (2 or 4)
    if (tid < ngrp) {
      float S = 0.f, Q = 0.f;
      for (int w = 0; w < 8; ++w)
        if (((w & 3) * 16) / Cg == tid) { S += wv_s[w]; Q += wv_q[w]; }
      int gg = m0 / Cg + tid;
      size_t pidx = (((size_t)b * 8 + gg) * gridDim.x + blockIdx.x) * 2;
      part[pidx] = S; part[pidx + 1] = Q;
    }
  }
}

// ------------------------- GN stats finalize -------------------------------
__global__ void stats_finalize_kernel(const float* __restrict__ part,
                                      float* __restrict__ stats, int nLblk, int n) {
  int bg = threadIdx.x;                    // <<<1,128>>> : (b,g) pairs
  float s = 0.f, q = 0.f;
  for (int i = 0; i < nLblk; ++i) {
    s += part[((size_t)bg * nLblk + i) * 2 + 0];
    q += part[((size_t)bg * nLblk + i) * 2 + 1];
  }
  float inv = 1.0f / (float)n;
  float mu = s * inv;
  float var = q * inv - mu * mu;
  stats[bg * 2 + 0] = mu;
  stats[bg * 2 + 1] = rsqrtf(var + 1e-5f);
}

// out = (res ? res[i] : 0) + [gelu]( (x-mu)*rstd*gamma + beta ).  res/out may alias.
__global__ __launch_bounds__(256) void gn_apply_kernel(
    const float* __restrict__ X, const float* res,
    const float* __restrict__ gamma, const float* __restrict__ beta,
    const float* __restrict__ stats, float* out, int C, int L, int Cg, int act) {
  size_t i = (size_t)blockIdx.x * 256 + threadIdx.x;   // over 16*C*L
  size_t CL = (size_t)C * L;
  int b = (int)(i / CL);
  int c = (int)((i - (size_t)b * CL) / L);
  int g = c / Cg;
  float mu = stats[(b * 8 + g) * 2 + 0];
  float rstd = stats[(b * 8 + g) * 2 + 1];
  float v = (X[i] - mu) * rstd * gamma[c] + beta[c];
  if (act == 1) v = geluf(v);
  if (res) v += res[i];
  out[i] = v;
}

// ---------------------------------------------------------------------------
// GRU recurrence. One workgroup, 768 threads = 24 waves; each wave owns two
// 16-row M-tiles of hg (768 gates).  Per step: hg = Whh(bf16) @ h + bhh via
// 8 K-frag WMMAs per tile, then gate math; h kept in LDS (f32 + bf16 mirror).
// xg layout (B,3H,T); ys layout (B,H,T).
// ---------------------------------------------------------------------------
#define GRU_HP 264   // padded bf16 row stride for h (528B, 16B-aligned)

__global__ __launch_bounds__(768) void gru_scan_kernel(
    const float* __restrict__ xg, const unsigned short* __restrict__ whh,
    const float* __restrict__ bhh, const float* __restrict__ h0,
    float* __restrict__ ys, int T) {
  extern __shared__ char smem[];
  unsigned short* h_bf = (unsigned short*)smem;          // [16][GRU_HP]
  float* h_f = (float*)(smem + 16 * GRU_HP * 2);         // [16][256]
  float* hg  = h_f + 16 * 256;                           // [768][16]

  const int tid = threadIdx.x;
  const int lane = tid & 31;
  const int wid = tid >> 5;                              // 0..23
  const int nlane = lane & 15;
  const int half8 = (lane & 16) ? 8 : 0;
  const int klo16 = (lane & 16);
  const int mhi = (lane & 16) ? 8 : 0;

  for (int i = tid; i < 16 * 256; i += 768) {
    float v = h0[i];
    int b = i >> 8, r = i & 255;
    h_f[b * 256 + r] = v;
    h_bf[b * GRU_HP + r] = f2bf(v);
  }
  __syncthreads();

  for (int t = 0; t < T; ++t) {
    // ---- hg = Whh @ h + bhh ----
#pragma unroll
    for (int tt = 0; tt < 2; ++tt) {
      const int tile = wid * 2 + tt;                     // 0..47
      const int m = tile * 16 + nlane;
      v8f acc = {};
      const unsigned short* wr = whh + (size_t)m * 256;
      const unsigned short* hr = h_bf + nlane * GRU_HP;
#pragma unroll
      for (int kf = 0; kf < 8; ++kf) {
        const int kb = kf * 32;
        v16bf afrag = cat16(ld8(wr + kb + half8), ld8(wr + kb + 16 + half8));
        v16bf bfrag = cat16(ld8(hr + kb + half8), ld8(hr + kb + 16 + half8));
        acc = __builtin_amdgcn_wmma_f32_16x16x32_bf16(false, afrag, false, bfrag,
                                                      (short)0, acc, false, false);
      }
      const int mbase = tile * 16 + mhi;
#pragma unroll
      for (int r = 0; r < 8; ++r) {
        int g = mbase + r;
        hg[g * 16 + nlane] = acc[r] + bhh[g];
      }
    }
    __syncthreads();
    // ---- gates + state update ----
    for (int i = tid; i < 16 * 256; i += 768) {
      int b = i >> 8, r = i & 255;
      size_t xb = ((size_t)b * 768) * T + t;
      float xr = xg[xb + (size_t)r * T];
      float xz = xg[xb + (size_t)(r + 256) * T];
      float xn = xg[xb + (size_t)(r + 512) * T];
      float rg = sigmf(xr + hg[r * 16 + b]);
      float zg = sigmf(xz + hg[(r + 256) * 16 + b]);
      float ng = tanhf(xn + rg * hg[(r + 512) * 16 + b]);
      float hnew = (1.0f - zg) * ng + zg * h_f[b * 256 + r];
      h_f[b * 256 + r] = hnew;
      h_bf[b * GRU_HP + r] = f2bf(hnew);
      ys[((size_t)b * 256 + r) * T + t] = hnew;
    }
    __syncthreads();
  }
}

// ------------------------- heads + additive synth --------------------------
__global__ __launch_bounds__(128) void heads_synth_kernel(
    const float* __restrict__ ys2,
    const float* __restrict__ note_w, const float* __restrict__ note_b,
    const float* __restrict__ vel_w,  const float* __restrict__ vel_b,
    const float* __restrict__ br_w,   const float* __restrict__ br_b,
    const float* __restrict__ no_w,   const float* __restrict__ no_b,
    const float* __restrict__ harmonic_logits,
    const float* __restrict__ decay_logits, const float* __restrict__ noise_decay_logits,
    const float* __restrict__ master_gain, const float* __restrict__ noise_gain,
    const float* __restrict__ amp, const float* __restrict__ phase,
    const float* __restrict__ noise, const float* __restrict__ prev_gate,
    const float* __restrict__ white_noise, float* __restrict__ out, int T) {
  const int b = blockIdx.x;
  const int tid = threadIdx.x;
  __shared__ float feat[256];
  __shared__ float heads[4][88];
  __shared__ float amp0s[88], noise0s[88], ndec[88], nsdec[88];
  __shared__ float harm[88][8];

  for (int c = tid; c < 256; c += 128)
    feat[c] = ys2[((size_t)b * 256 + c) * T + (T - 1)];
  __syncthreads();

  for (int i = tid; i < 4 * 88; i += 128) {
    int h = i / 88, n = i - h * 88;
    const float* w = (h == 0 ? note_w : h == 1 ? vel_w : h == 2 ? br_w : no_w) + (size_t)n * 256;
    const float* bb = (h == 0 ? note_b : h == 1 ? vel_b : h == 2 ? br_b : no_b);
    float s = bb[n];
    for (int c = 0; c < 256; ++c) s += feat[c] * w[c];
    heads[h][n] = s;
  }
  __syncthreads();

  if (tid < 88) {
    int n = tid;
    float gate = sigmf(heads[0][n]);
    float vel  = sigmf(heads[1][n]);
    float br   = sigmf(heads[2][n]);
    float namt = sigmf(heads[3][n]);
    float onset = fmaxf(gate - prev_gate[b * 88 + n], 0.0f);
    float nd  = 0.9995f + 0.00049f * sigmf(decay_logits[n]);
    float nsd = 0.9f + 0.09f * sigmf(noise_decay_logits[n]);
    ndec[n] = nd; nsdec[n] = nsd;
    amp0s[n]   = fmaxf(amp[b * 88 + n] * nd, gate * vel);
    noise0s[n] = noise[b * 88 + n] * nsd + onset * namt;
    float mx = -1e30f;
    for (int p = 0; p < 8; ++p) mx = fmaxf(mx, harmonic_logits[n * 8 + p]);
    float se = 0.f, hv[8];
    for (int p = 0; p < 8; ++p) { hv[p] = expf(harmonic_logits[n * 8 + p] - mx); se += hv[p]; }
    float tiltc = 1.5f - 1.2f * br, sum = 0.f;
    for (int p = 0; p < 8; ++p) { hv[p] = (hv[p] / se) * expf(-(float)(p + 1) * tiltc); sum += hv[p]; }
    float inv = 1.0f / (sum + 1e-8f);
    for (int p = 0; p < 8; ++p) harm[n][p] = hv[p] * inv;
  }
  __syncthreads();

  if (tid < 96) {
    const int t = tid;
    const float tf = (float)(t + 1);
    float tonal = 0.f, noisy = 0.f;
    for (int n = 0; n < 88; ++n) {
      float f0 = 440.0f * exp2f((float)(21 + n - 69) * (1.0f / 12.0f));
      float a = amp0s[n] * powf(ndec[n], (float)t);
      for (int p = 0; p < 8; ++p) {
        float inc = (2.0f * 3.14159265358979323846f / 48000.0f) * f0 * (float)(p + 1);
        float ph = phase[((size_t)b * 88 + n) * 8 + p] + inc * tf;
        tonal += sinf(ph) * harm[n][p] * a;
      }
      noisy += white_noise[((size_t)b * 88 + n) * 96 + t] * noise0s[n] * powf(nsdec[n], (float)t);
    }
    out[b * 96 + t] = master_gain[0] * tonal + noise_gain[0] * noisy;
  }
}

// ---------------------------------------------------------------------------
// Host orchestration.
// Input order: top-level setup_inputs() insertion order; 'params' flattened
// JAX-style (recursively sorted dict keys) -> 141 leaves total.
// ---------------------------------------------------------------------------
enum {
  IN_XCTX = 0, IN_ONSET, IN_H0, IN_AMP, IN_PHASE, IN_NOISE, IN_PREVGATE, IN_WHITE,
  P_BR_B = 8, P_BR_W = 9, P_DECAY = 10,
  E1 = 11, E2 = 37, E3 = 63,               // each stage: down_b, down_w, r1[8], r2[8], r3[8]
  P_GRU0 = 89, P_GRU1 = 93,                // b_hh, b_ih, w_hh, w_ih
  P_HARM = 97, P_IN_B = 98, P_IN_W = 99, P_MASTER = 100,
  MID0 = 101, MID1 = 109, MID2 = 117, MID3 = 125,   // resp: c1_b,c1_w,c2_b,c2_w,n1_b,n1_g,n2_b,n2_g
  P_NO_B = 133, P_NO_W = 134, P_NOISE_DECAY = 135, P_NOISE_GAIN = 136,
  P_NOTE_B = 137, P_NOTE_W = 138, P_VEL_B = 139, P_VEL_W = 140
};

extern "C" void kernel_launch(void* const* d_in, const int* in_sizes, int n_in,
                              void* d_out, int out_size, void* d_ws, size_t ws_size,
                              hipStream_t stream) {
  const int NB = 16;
  auto FP = [&](int i) { return (const float*)d_in[i]; };

  // ---- workspace carve-up (identical layout every call: deterministic) ----
  char* ws = (char*)d_ws;
  size_t off = 0;
  auto alloc = [&](size_t bytes) {
    off = (off + 255) & ~(size_t)255;
    void* p = ws + off;
    off += bytes;
    return p;
  };
  const size_t BIG = (size_t)16 * 64 * 8192;   // max intermediate elems (8.39M)
  float* xin = (float*)alloc((size_t)16 * 2 * 8192 * 4);
  float* P0  = (float*)alloc(BIG * 4);
  float* P1  = (float*)alloc(BIG * 4);
  float* P2  = (float*)alloc(BIG * 4);
  float* XG  = (float*)alloc((size_t)16 * 768 * 1024 * 4);
  float* stats = (float*)alloc(128 * 2 * 4);
  float* part  = (float*)alloc((size_t)128 * 64 * 2 * 4);
  unsigned short* whh0 = (unsigned short*)alloc((size_t)768 * 256 * 2);
  unsigned short* whh1 = (unsigned short*)alloc((size_t)768 * 256 * 2);
  (void)ws_size; (void)n_in; (void)in_sizes; (void)out_size;

  // ---- launch helpers ----
  auto conv = [&](const float* X, int iw, int ib, float* Y, float* prt,
                  int Cin, int Cout, int Lout, int Lin, int k, int stride,
                  int dil, int act) {
    int welems = Cout * Cin * k;
    unsigned short* wbf = (unsigned short*)alloc((size_t)welems * 2);
    cvt_bf16_kernel<<<(welems + 255) / 256, 256, 0, stream>>>(FP(iw), wbf, welems);
    dim3 g(Lout / 128, Cout / 64, NB);
    conv_wmma_kernel<<<g, 256, 0, stream>>>(X, wbf, FP(ib), Y, prt, Cin, Cout,
                                            Lout, Lin, k, stride, dil, act, Cout / 8);
  };
  auto gnapply = [&](const float* Xin_, const float* res, float* Y, int ig, int ib,
                     int C, int L, int act) {
    stats_finalize_kernel<<<1, 128, 0, stream>>>(part, stats, L / 128, (C / 8) * L);
    size_t total = (size_t)16 * C * L;
    gn_apply_kernel<<<(unsigned)(total / 256), 256, 0, stream>>>(
        Xin_, res, FP(ig), FP(ib), stats, Y, C, L, C / 8, act);
  };
  // resp leaves: +0 c1_b, +1 c1_w, +2 c2_b, +3 c2_w, +4 n1_b, +5 n1_g, +6 n2_b, +7 n2_g
  auto resb = [&](float* X, float* T1, float* T2, int rb, int C, int L, int dil) {
    conv(X, rb + 1, rb + 0, T1, part, C, C, L, L, 5, 1, dil, 0);
    gnapply(T1, nullptr, T2, rb + 5, rb + 4, C, L, 1);
    conv(T2, rb + 3, rb + 2, T1, part, C, C, L, L, 5, 1, 1, 0);
    gnapply(T1, X, X, rb + 7, rb + 6, C, L, 1);         // X += gelu(gn(T1))
  };
  auto stage = [&](float* Xin_, float* Xout, float* T1, float* T2, int sb,
                   int Cin, int Cout, int Lout) {
    conv(Xin_, sb + 1, sb + 0, Xout, nullptr, Cin, Cout, Lout, Lout * 2, 7, 2, 1, 1);
    resb(Xout, T1, T2, sb + 2, Cout, Lout, 1);
    resb(Xout, T1, T2, sb + 10, Cout, Lout, 2);
    resb(Xout, T1, T2, sb + 18, Cout, Lout, 4);
  };

  // ---- front end ----
  concat2_kernel<<<(16 * 8192) / 256, 256, 0, stream>>>(FP(IN_XCTX), FP(IN_ONSET), xin, 8192);
  conv(xin, P_IN_W, P_IN_B, P0, nullptr, 2, 64, 8192, 8192, 9, 1, 1, 1);   // z0

  // ---- encoder ----
  stage(P0, P1, P2, P0, E1, 64, 128, 4096);    // X=P1
  stage(P1, P2, P0, P1, E2, 128, 256, 2048);   // X=P2
  stage(P2, P0, P1, P2, E3, 256, 256, 1024);   // X=P0

  // ---- mid blocks ----
  resb(P0, P1, P2, MID0, 256, 1024, 1);
  resb(P0, P1, P2, MID1, 256, 1024, 2);
  resb(P0, P1, P2, MID2, 256, 1024, 4);
  resb(P0, P1, P2, MID3, 256, 1024, 8);

  // ---- GRU stack ----
  const size_t gru_smem = 16 * GRU_HP * 2 + 16 * 256 * 4 + 768 * 16 * 4;
  cvt_bf16_kernel<<<(768 * 256) / 256, 256, 0, stream>>>(FP(P_GRU0 + 2), whh0, 768 * 256);
  conv(P0, P_GRU0 + 3, P_GRU0 + 1, XG, nullptr, 256, 768, 1024, 1024, 1, 1, 1, 0);
  gru_scan_kernel<<<1, 768, gru_smem, stream>>>(XG, whh0, FP(P_GRU0 + 0),
                                                FP(IN_H0), P1, 1024);
  cvt_bf16_kernel<<<(768 * 256) / 256, 256, 0, stream>>>(FP(P_GRU1 + 2), whh1, 768 * 256);
  conv(P1, P_GRU1 + 3, P_GRU1 + 1, XG, nullptr, 256, 768, 1024, 1024, 1, 1, 1, 0);
  gru_scan_kernel<<<1, 768, gru_smem, stream>>>(XG, whh1, FP(P_GRU1 + 0),
                                                FP(IN_H0) + 16 * 256, P2, 1024);

  // ---- heads + synthesis ----
  heads_synth_kernel<<<16, 128, 0, stream>>>(
      P2, FP(P_NOTE_W), FP(P_NOTE_B), FP(P_VEL_W), FP(P_VEL_B),
      FP(P_BR_W), FP(P_BR_B), FP(P_NO_W), FP(P_NO_B),
      FP(P_HARM), FP(P_DECAY), FP(P_NOISE_DECAY),
      FP(P_MASTER), FP(P_NOISE_GAIN),
      FP(IN_AMP), FP(IN_PHASE), FP(IN_NOISE), FP(IN_PREVGATE), FP(IN_WHITE),
      (float*)d_out, 1024);
}
